// Model_27685359190754
// MI455X (gfx1250) — matrix operand order repaired and checked by reading, not compiled, not run on hardware
//
#include <hip/hip_runtime.h>

typedef __bf16 bf16_t;
typedef __attribute__((ext_vector_type(8)))  bf16_t v8bf;
typedef __attribute__((ext_vector_type(16))) bf16_t v16bf;
typedef __attribute__((ext_vector_type(8)))  float  v8f;

#define TOPN 1000
#define NDET 100
#define NCLS 80
#define NBINS 65536
#define BOX_BATCH 196416

// ---------------- constant tables ----------------
__constant__ float c_anchors[5][9][4] = {
 {{-18.0f,-8.0f,25.0f,15.0f},{-23.7183f,-11.1191f,30.7183f,18.1191f},{-30.9228f,-15.0488f,37.9228f,22.0488f},
  {-12.0f,-12.0f,19.0f,19.0f},{-16.1587f,-16.1587f,23.1587f,23.1587f},{-21.3984f,-21.3984f,28.3984f,28.3984f},
  {-8.0f,-20.0f,15.0f,27.0f},{-11.1191f,-26.2381f,18.1191f,33.2381f},{-15.0488f,-34.0976f,22.0488f,41.0976f}},
 {{-38.0f,-16.0f,53.0f,31.0f},{-49.9564f,-22.2381f,64.9564f,37.2381f},{-65.0204f,-30.0976f,80.0204f,45.0976f},
  {-24.0f,-24.0f,39.0f,39.0f},{-32.3175f,-32.3175f,47.3175f,47.3175f},{-42.7968f,-42.7968f,57.7968f,57.7968f},
  {-14.0f,-36.0f,29.0f,51.0f},{-19.7183f,-47.4365f,34.7183f,62.4365f},{-26.9228f,-61.8456f,41.9228f,76.8456f}},
 {{-74.0f,-28.0f,105.0f,59.0f},{-97.3929f,-39.4365f,128.3929f,70.4365f},{-126.8661f,-53.8456f,157.8661f,84.8456f},
  {-48.0f,-48.0f,79.0f,79.0f},{-64.6349f,-64.6349f,95.6349f,95.6349f},{-85.5937f,-85.5937f,116.5937f,116.5937f},
  {-30.0f,-76.0f,61.0f,107.0f},{-41.9564f,-99.9127f,72.9564f,130.9127f},{-57.0204f,-130.0409f,88.0204f,161.0409f}},
 {{-150.0f,-60.0f,213.0f,123.0f},{-197.3056f,-83.9127f,260.3056f,146.9127f},{-256.907f,-114.0409f,319.907f,177.0409f},
  {-96.0f,-96.0f,159.0f,159.0f},{-129.2699f,-129.2699f,192.2699f,192.2699f},{-171.1873f,-171.1873f,234.1873f,234.1873f},
  {-58.0f,-148.0f,121.0f,211.0f},{-81.3929f,-194.7858f,144.3929f,257.7858f},{-110.8661f,-253.7322f,173.8661f,316.7322f}},
 {{-298.0f,-116.0f,425.0f,243.0f},{-392.0914f,-162.7858f,519.0914f,289.7858f},{-510.6392f,-221.7322f,637.6392f,348.7322f},
  {-192.0f,-192.0f,319.0f,319.0f},{-258.5398f,-258.5398f,385.5398f,385.5398f},{-342.3747f,-342.3747f,469.3747f,469.3747f},
  {-118.0f,-300.0f,245.0f,427.0f},{-165.3056f,-394.6113f,292.3056f,521.6113f},{-224.907f,-513.814f,351.907f,640.814f}}};

__constant__ int   c_HW[5]     = {4096, 1024, 256, 64, 16};
__constant__ int   c_Wd[5]     = {64, 32, 16, 8, 4};
__constant__ float c_stride[5] = {8.f, 16.f, 32.f, 64.f, 128.f};
__constant__ int   c_boxOff[5] = {0, 147456, 184320, 193536, 195840};
static const int   h_boxOff[5] = {0, 147456, 184320, 193536, 195840};

// ---------------- conversion kernels ----------------
__global__ void feat_to_bf16(const float* __restrict__ F, bf16_t* __restrict__ Xp, int H, int W) {
  int Wp = W + 2;
  int total = (H + 2) * Wp * 256;
  for (int t = blockIdx.x * blockDim.x + threadIdx.x; t < total; t += gridDim.x * blockDim.x) {
    int p = t >> 8;
    int c = t & 255;
    int yy = p / Wp, xx = p % Wp;
    float v = 0.f;
    if (yy >= 1 && yy <= H && xx >= 1 && xx <= W)
      v = F[(size_t)c * H * W + (size_t)(yy - 1) * W + (xx - 1)];
    Xp[t] = (bf16_t)v;
  }
}

__global__ void weight_to_bf16(const float* __restrict__ Ws, bf16_t* __restrict__ Wd, int Cout, int CoutPad) {
  int total = CoutPad * 2304;
  for (int t = blockIdx.x * blockDim.x + threadIdx.x; t < total; t += gridDim.x * blockDim.x) {
    int m = t / 2304, k = t % 2304;
    int kk = k / 256, c = k % 256;
    int ky = kk / 3, kx = kk % 3;
    float v = 0.f;
    if (m < Cout) v = Ws[(((size_t)m * 256 + c) * 3 + ky) * 3 + kx];
    Wd[t] = (bf16_t)v;
  }
}

// ---------------- WMMA implicit-GEMM conv3x3, 2x2 register blocking ----------------
__device__ __forceinline__ v16bf loadA(const bf16_t* p) {
  v8bf lo = *(const v8bf*)(p);
  v8bf hi = *(const v8bf*)(p + 8);
  return __builtin_shufflevector(lo, hi, 0,1,2,3,4,5,6,7,8,9,10,11,12,13,14,15);
}

__global__ __launch_bounds__(128)
void conv3x3_wmma(const bf16_t* __restrict__ X, const bf16_t* __restrict__ Wt,
                  const float* __restrict__ bias, float* __restrict__ OutF,
                  bf16_t* __restrict__ OutB, int H, int W, int Cout, int CoutPad, int mode) {
  const int lane   = threadIdx.x & 31;
  const int wave   = threadIdx.x >> 5;
  const int laneHi = lane >> 4;
  const int lm     = lane & 15;
  const int N      = H * W;
  const int n0     = (blockIdx.y * 4 + wave) * 32;          // 2 N-tiles of 16 per wave
  if (n0 >= N) return;                                      // wave-uniform: EXEC all-ones at WMMA
  const bool v2 = (n0 + 16) < N;                            // second N-tile valid (wave-uniform)
  const int Wp = W + 2;

  // N-tile 0 column for this lane
  const int na = n0 + lm;
  const int y0 = na / W, x0 = na % W;
  // N-tile 1 column, clamped to a safe pixel when the tile is partial/invalid
  int nb = n0 + 16 + lm; if (nb >= N) nb = N - 1;
  const int y1 = nb / W, x1 = nb % W;

  // A rows: two M-tiles of 16
  const int mt = blockIdx.x;                                // 32 output channels per block-x
  const bf16_t* Arow0 = Wt + (size_t)(mt * 32 + lm)      * 2304 + laneHi * 16;
  const bf16_t* Arow1 = Wt + (size_t)(mt * 32 + 16 + lm) * 2304 + laneHi * 16;

  v8f acc00 = {}, acc01 = {}, acc10 = {}, acc11 = {};
  for (int kk = 0; kk < 9; ++kk) {
    const int ky = kk / 3, kx = kk % 3;
    const bf16_t* B0 = X + ((size_t)(y0 + ky) * Wp + (x0 + kx)) * 256 + laneHi * 16;
    const bf16_t* B1 = X + ((size_t)(y1 + ky) * Wp + (x1 + kx)) * 256 + laneHi * 16;
    const bf16_t* A0 = Arow0 + kk * 256;
    const bf16_t* A1 = Arow1 + kk * 256;
#pragma unroll
    for (int c0 = 0; c0 < 256; c0 += 32) {
      v16bf a0 = loadA(A0 + c0);
      v16bf a1 = loadA(A1 + c0);
      v16bf b0 = *(const v16bf*)(B0 + c0);
      v16bf b1 = *(const v16bf*)(B1 + c0);
      acc00 = __builtin_amdgcn_wmma_f32_16x16x32_bf16(false, a0, false, b0, (short)0, acc00, false, false);
      acc01 = __builtin_amdgcn_wmma_f32_16x16x32_bf16(false, a0, false, b1, (short)0, acc01, false, false);
      acc10 = __builtin_amdgcn_wmma_f32_16x16x32_bf16(false, a1, false, b0, (short)0, acc10, false, false);
      acc11 = __builtin_amdgcn_wmma_f32_16x16x32_bf16(false, a1, false, b1, (short)0, acc11, false, false);
    }
  }

  // epilogue: D tile (mi,ni): M = mi*16 + laneHi*8 + r, N-column = n0 + ni*16 + lm
  v8f accs[2][2] = {{acc00, acc01}, {acc10, acc11}};
#pragma unroll
  for (int ni = 0; ni < 2; ++ni) {
    if (ni == 1 && !v2) break;                              // wave-uniform
    const int n  = n0 + ni * 16 + lm;
    const int yy = n / W, xx = n % W;
#pragma unroll
    for (int mi = 0; mi < 2; ++mi) {
      const int ch0 = mt * 32 + mi * 16 + laneHi * 8;
      if (mode == 0) {
        const int pidx = (yy + 1) * Wp + (xx + 1);
        v8bf o;
#pragma unroll
        for (int r = 0; r < 8; ++r) {
          float v = accs[mi][ni][r] + bias[ch0 + r];
          o[r] = (bf16_t)(v > 0.f ? v : 0.f);
        }
        *(v8bf*)(OutB + (size_t)pidx * 256 + ch0) = o;
      } else {
#pragma unroll
        for (int r = 0; r < 8; ++r) {
          int ch = ch0 + r;
          if (ch < Cout) {
            float v = accs[mi][ni][r] + bias[ch];
            if (mode == 1) v = 1.f / (1.f + __expf(-v));
            OutF[(size_t)ch * N + n] = v;
          }
        }
      }
    }
  }
}

// ---------------- top-k selection ----------------
__device__ __forceinline__ unsigned score_key(float f) {
  unsigned u = __float_as_uint(f);
  return (u & 0x80000000u) ? ~u : (u | 0x80000000u);
}

__global__ void hist_kernel(const float* __restrict__ S, int n, unsigned* __restrict__ hist) {
  for (int i = blockIdx.x * blockDim.x + threadIdx.x; i < n; i += gridDim.x * blockDim.x)
    atomicAdd(&hist[score_key(S[i]) >> 16], 1u);
}

__global__ void select_cutoff_kernel(const unsigned* __restrict__ hist, int2* __restrict__ cut) {
  const unsigned* h = hist + (size_t)blockIdx.x * NBINS;
  __shared__ unsigned csum[1024];
  unsigned s = 0;
  int base = threadIdx.x * 64;
  for (int i = 0; i < 64; ++i) s += h[base + i];
  csum[threadIdx.x] = s;
  __syncthreads();
  if (threadIdx.x == 0) {
    unsigned acc = 0; int cutoff = -1; unsigned nAbove = 0;
    for (int t = 1023; t >= 0; --t) {
      if (acc + csum[t] >= TOPN) {
        for (int bin = t * 64 + 63; bin >= t * 64; --bin) {
          unsigned hb = h[bin];
          if (acc + hb >= TOPN) { cutoff = bin; nAbove = acc; break; }
          acc += hb;
        }
        break;
      }
      acc += csum[t];
    }
    cut[blockIdx.x].x = cutoff;
    cut[blockIdx.x].y = (int)nAbove;
  }
}

__global__ void compact_kernel(const float* __restrict__ S, int n, const int2* __restrict__ cut,
                               int* __restrict__ cnt, float* __restrict__ cS, int* __restrict__ cI) {
  int cutoff = cut->x, nAbove = cut->y;
  for (int i = blockIdx.x * blockDim.x + threadIdx.x; i < n; i += gridDim.x * blockDim.x) {
    float s = S[i];
    int bin = (int)(score_key(s) >> 16);
    if (bin > cutoff) {
      int p = atomicAdd(&cnt[0], 1);
      if (p < TOPN) { cS[p] = s; cI[p] = i; }
    } else if (bin == cutoff) {
      int t = atomicAdd(&cnt[1], 1);
      if (t < TOPN - nAbove) { cS[nAbove + t] = s; cI[nAbove + t] = i; }
    }
  }
}

// ---------------- decode ----------------
__global__ void decode_kernel(float* __restrict__ cS, const int* __restrict__ cI,
                              const float* __restrict__ boxAll,
                              float* __restrict__ cB, float* __restrict__ cC) {
  int slot = blockIdx.x * blockDim.x + threadIdx.x;
  if (slot >= 2 * 5 * TOPN) return;
  int b   = slot / (5 * TOPN);
  int lvl = (slot / TOPN) % 5;
  float s = cS[slot];
  int idx = cI[slot];
  int Wd = c_Wd[lvl], HW = c_HW[lvl];
  int x  = idx % Wd;
  int y  = (idx / Wd) % Wd;
  int ch = idx / HW;
  int cls = ch % NCLS;
  int a   = ch / NCLS;
  const float* bo = boxAll + (size_t)b * BOX_BATCH + c_boxOff[lvl];
  int pix = y * Wd + x;
  float d0 = bo[(a * 4 + 0) * HW + pix];
  float d1 = bo[(a * 4 + 1) * HW + pix];
  float d2 = bo[(a * 4 + 2) * HW + pix];
  float d3 = bo[(a * 4 + 3) * HW + pix];
  float st = c_stride[lvl];
  float g0 = x * st + c_anchors[lvl][a][0];
  float g1 = y * st + c_anchors[lvl][a][1];
  float g2 = x * st + c_anchors[lvl][a][2];
  float g3 = y * st + c_anchors[lvl][a][3];
  float aw = g2 - g0 + 1.f, ah = g3 - g1 + 1.f;
  float cx = g0 + 0.5f * aw, cy = g1 + 0.5f * ah;
  float px = d0 * aw + cx,   py = d1 * ah + cy;
  float pw = __expf(d2) * aw, ph = __expf(d3) * ah;
  float M  = Wd * st - 1.f;
  float x1 = fminf(fmaxf(px - 0.5f * pw, 0.f), M);
  float y1 = fminf(fmaxf(py - 0.5f * ph, 0.f), M);
  float x2 = fminf(fmaxf(px + 0.5f * pw - 1.f, 0.f), M);
  float y2 = fminf(fmaxf(py + 0.5f * ph - 1.f, 0.f), M);
  bool valid = s >= 0.05f;
  cS[slot]       = valid ? s  : 0.f;
  cB[slot*4 + 0] = valid ? x1 : 0.f;
  cB[slot*4 + 1] = valid ? y1 : 0.f;
  cB[slot*4 + 2] = valid ? x2 : 0.f;
  cB[slot*4 + 3] = valid ? y2 : 0.f;
  cC[slot]       = valid ? (float)cls : 0.f;
}

// ---------------- NMS ----------------
#define NMSB 256
__global__ void nms_kernel(const float* __restrict__ cS, const float* __restrict__ cB,
                           const float* __restrict__ cC, int* __restrict__ valid,
                           float* __restrict__ out) {
  const int b  = blockIdx.x;
  const int NC = 5 * TOPN;
  const float* S  = cS + (size_t)b * NC;
  const float* Bx = cB + (size_t)b * NC * 4;
  const float* C  = cC + (size_t)b * NC;
  int* V = valid + (size_t)b * NC;
  for (int i = threadIdx.x; i < NC; i += blockDim.x) V[i] = S[i] > 0.f ? 1 : 0;
  __shared__ float rs[NMSB];
  __shared__ int   ri[NMSB];
  __shared__ int   sBest;
  __syncthreads();
  for (int it = 0; it < NDET; ++it) {
    float bs = -1.f; int bi = -1;
    for (int i = threadIdx.x; i < NC; i += blockDim.x)
      if (V[i]) { float s = S[i]; if (s > bs || (s == bs && (bi < 0 || i < bi))) { bs = s; bi = i; } }
    rs[threadIdx.x] = bs; ri[threadIdx.x] = bi;
    __syncthreads();
    for (int off = NMSB / 2; off > 0; off >>= 1) {
      if (threadIdx.x < off) {
        float so = rs[threadIdx.x + off]; int io = ri[threadIdx.x + off];
        if (io >= 0 && (so > rs[threadIdx.x] || (so == rs[threadIdx.x] && (ri[threadIdx.x] < 0 || io < ri[threadIdx.x])))) {
          rs[threadIdx.x] = so; ri[threadIdx.x] = io;
        }
      }
      __syncthreads();
    }
    if (threadIdx.x == 0) sBest = ri[0];
    __syncthreads();
    int best = sBest;
    bool any = best >= 0;
    float si = 0.f, ci = 0.f, b0 = 0.f, b1 = 0.f, b2 = 0.f, b3 = 0.f;
    if (any) { si = S[best]; ci = C[best]; b0 = Bx[best*4]; b1 = Bx[best*4+1]; b2 = Bx[best*4+2]; b3 = Bx[best*4+3]; }
    if (threadIdx.x == 0) {
      out[b * NDET + it] = any ? si : 0.f;
      float* ob = out + 2 * NDET + ((size_t)b * NDET + it) * 4;
      ob[0] = b0; ob[1] = b1; ob[2] = b2; ob[3] = b3;
      out[2 * NDET + 2 * NDET * 4 + b * NDET + it] = any ? ci : 0.f;
    }
    if (any) {
      float ai = (b2 - b0 + 1.f) * (b3 - b1 + 1.f);
      for (int i = threadIdx.x; i < NC; i += blockDim.x) {
        if (!V[i]) continue;
        float e0 = Bx[i*4], e1 = Bx[i*4+1], e2 = Bx[i*4+2], e3 = Bx[i*4+3];
        float iw = fmaxf(fminf(e2, b2) - fmaxf(e0, b0) + 1.f, 0.f);
        float ih = fmaxf(fminf(e3, b3) - fmaxf(e1, b1) + 1.f, 0.f);
        float inter = iw * ih;
        float area  = (e2 - e0 + 1.f) * (e3 - e1 + 1.f);
        float iou   = inter / (area + ai - inter);
        if (S[i] <= si && iou > 0.5f && C[i] == ci) V[i] = 0;
      }
    }
    __syncthreads();
  }
}

// ---------------- host driver ----------------
extern "C" void kernel_launch(void* const* d_in, const int* in_sizes, int n_in,
                              void* d_out, int out_size, void* d_ws, size_t ws_size,
                              hipStream_t stream) {
  (void)in_sizes; (void)n_in; (void)out_size; (void)ws_size;
  char* ws = (char*)d_ws;
  size_t off = 0;
  auto alloc = [&](size_t bytes) -> void* {
    void* p = ws + off; off = (off + bytes + 255) & ~(size_t)255; return p;
  };

  const int clsCout[5] = {256, 256, 256, 256, 720};
  const int clsPad [5] = {256, 256, 256, 256, 736};   // pad to multiple of 32 for 2x2 blocking
  const int boxCout[5] = {256, 256, 256, 256, 36};
  const int boxPad [5] = {256, 256, 256, 256, 64};
  bf16_t* wCls[5]; bf16_t* wBox[5];
  for (int i = 0; i < 5; ++i) wCls[i] = (bf16_t*)alloc((size_t)clsPad[i] * 2304 * 2);
  for (int i = 0; i < 5; ++i) wBox[i] = (bf16_t*)alloc((size_t)boxPad[i] * 2304 * 2);

  const size_t ACT = (size_t)66 * 66 * 256;
  bf16_t* bufIn = (bf16_t*)alloc(ACT * 2);
  bf16_t* bufA  = (bf16_t*)alloc(ACT * 2);
  bf16_t* bufB  = (bf16_t*)alloc(ACT * 2);

  const size_t clsOffH[5] = {0, 2949120, 3686400, 3870720, 3916800};
  const size_t CLS_BATCH  = 3928320;
  float* clsOut = (float*)alloc(2 * CLS_BATCH * 4);
  float* boxOut = (float*)alloc(2 * (size_t)BOX_BATCH * 4);

  unsigned* hist  = (unsigned*)alloc((size_t)10 * NBINS * 4);
  int2*     cuts  = (int2*)alloc(10 * sizeof(int2));
  int*      cnts  = (int*)alloc(10 * 2 * 4);
  float*    candS = (float*)alloc(2 * 5 * TOPN * 4);
  int*      candI = (int*)alloc(2 * 5 * TOPN * 4);
  float*    candB = (float*)alloc(2 * 5 * TOPN * 4 * 4);
  float*    candC = (float*)alloc(2 * 5 * TOPN * 4);
  int*      validB = (int*)alloc(2 * 5 * TOPN * 4);

  hipMemsetAsync(hist, 0, (size_t)10 * NBINS * 4, stream);
  hipMemsetAsync(cnts, 0, 10 * 2 * 4, stream);
  hipMemsetAsync(candS, 0, 2 * 5 * TOPN * 4, stream);
  hipMemsetAsync(candI, 0, 2 * 5 * TOPN * 4, stream);

  for (int i = 0; i < 5; ++i) {
    int tc = clsPad[i] * 2304;
    weight_to_bf16<<<(tc + 255) / 256, 256, 0, stream>>>((const float*)d_in[5 + 4 * i], wCls[i], clsCout[i], clsPad[i]);
    int tb = boxPad[i] * 2304;
    weight_to_bf16<<<(tb + 255) / 256, 256, 0, stream>>>((const float*)d_in[7 + 4 * i], wBox[i], boxCout[i], boxPad[i]);
  }

  const int Hs[5] = {64, 32, 16, 8, 4};
  for (int b = 0; b < 2; ++b) {
    for (int l = 0; l < 5; ++l) {
      int H = Hs[l], W = H, HW = H * W, P = (H + 2) * (W + 2);
      hipMemsetAsync(bufA, 0, (size_t)P * 256 * 2, stream);
      hipMemsetAsync(bufB, 0, (size_t)P * 256 * 2, stream);
      feat_to_bf16<<<(P * 256 + 255) / 256, 256, 0, stream>>>(
          (const float*)d_in[l] + (size_t)b * 256 * HW, bufIn, H, W);

      dim3 blk(128);
      dim3 g256(256 / 32, (HW + 127) / 128);
      float* co = clsOut + (size_t)b * CLS_BATCH + clsOffH[l];
      float* bo = boxOut + (size_t)b * BOX_BATCH + h_boxOff[l];
      conv3x3_wmma<<<g256, blk, 0, stream>>>(bufIn, wCls[0], (const float*)d_in[6],  nullptr, bufA, H, W, 256, 256, 0);
      conv3x3_wmma<<<g256, blk, 0, stream>>>(bufA,  wCls[1], (const float*)d_in[10], nullptr, bufB, H, W, 256, 256, 0);
      conv3x3_wmma<<<g256, blk, 0, stream>>>(bufB,  wCls[2], (const float*)d_in[14], nullptr, bufA, H, W, 256, 256, 0);
      conv3x3_wmma<<<g256, blk, 0, stream>>>(bufA,  wCls[3], (const float*)d_in[18], nullptr, bufB, H, W, 256, 256, 0);
      dim3 gcls(736 / 32, (HW + 127) / 128);
      conv3x3_wmma<<<gcls, blk, 0, stream>>>(bufB,  wCls[4], (const float*)d_in[22], co, nullptr, H, W, 720, 736, 1);
      conv3x3_wmma<<<g256, blk, 0, stream>>>(bufIn, wBox[0], (const float*)d_in[8],  nullptr, bufA, H, W, 256, 256, 0);
      conv3x3_wmma<<<g256, blk, 0, stream>>>(bufA,  wBox[1], (const float*)d_in[12], nullptr, bufB, H, W, 256, 256, 0);
      conv3x3_wmma<<<g256, blk, 0, stream>>>(bufB,  wBox[2], (const float*)d_in[16], nullptr, bufA, H, W, 256, 256, 0);
      conv3x3_wmma<<<g256, blk, 0, stream>>>(bufA,  wBox[3], (const float*)d_in[20], nullptr, bufB, H, W, 256, 256, 0);
      dim3 gbox(64 / 32, (HW + 127) / 128);
      conv3x3_wmma<<<gbox, blk, 0, stream>>>(bufB,  wBox[4], (const float*)d_in[24], bo, nullptr, H, W, 36, 64, 2);
    }
  }

  for (int b = 0; b < 2; ++b)
    for (int l = 0; l < 5; ++l) {
      int HW = Hs[l] * Hs[l];
      int n = 720 * HW;
      const float* sc = clsOut + (size_t)b * CLS_BATCH + clsOffH[l];
      int nb = (n + 255) / 256; if (nb > 4096) nb = 4096;
      hist_kernel<<<nb, 256, 0, stream>>>(sc, n, hist + (size_t)(b * 5 + l) * NBINS);
    }
  select_cutoff_kernel<<<10, 1024, 0, stream>>>(hist, cuts);
  for (int b = 0; b < 2; ++b)
    for (int l = 0; l < 5; ++l) {
      int HW = Hs[l] * Hs[l];
      int n = 720 * HW;
      const float* sc = clsOut + (size_t)b * CLS_BATCH + clsOffH[l];
      int nb = (n + 255) / 256; if (nb > 4096) nb = 4096;
      int k = b * 5 + l;
      compact_kernel<<<nb, 256, 0, stream>>>(sc, n, cuts + k, cnts + 2 * k,
                                             candS + (size_t)k * TOPN, candI + (size_t)k * TOPN);
    }

  decode_kernel<<<(2 * 5 * TOPN + 255) / 256, 256, 0, stream>>>(candS, candI, boxOut, candB, candC);
  nms_kernel<<<2, NMSB, 0, stream>>>(candS, candB, candC, validB, (float*)d_out);
}
